// Phi3Mamba_71433896067317
// MI455X (gfx1250) — matrix operand order repaired and checked by reading, not compiled, not run on hardware
//
#include <hip/hip_runtime.h>
#include <hip/hip_bf16.h>
#include <math.h>

#define D_MODEL 4096
#define D_INNER 8192
#define D_STATE 16
#define D_CONV  4
#define DT_RANK 256
#define SEQ     2048
#define NSSM    (DT_RANK + 2 * D_STATE)   // 288

typedef __attribute__((ext_vector_type(16))) __bf16 v16bf;
typedef __attribute__((ext_vector_type(8)))  __bf16 v8bf;
typedef __attribute__((ext_vector_type(8)))  float  v8f;

#define TILE_M 128
#define TILE_N 128
#define TILE_K 32
#define LDSS   40                      // bf16 row stride: 80B, 16B-aligned chunks
#define TBUF_BYTES (TILE_M * LDSS * 2) // one LDS tile buffer: 10240 B

__device__ __forceinline__ float softplus_f(float x) {
    return (x > 20.f) ? x : log1pf(__expf(x));
}
__device__ __forceinline__ float silu_f(float x) {
    return x * (1.f / (1.f + __expf(-x)));
}

// CDNA5 async copy: 16 bytes global -> LDS, tracked by ASYNCcnt.
__device__ __forceinline__ void async_cp16(unsigned lds_off, const __bf16* g) {
    asm volatile("global_load_async_to_lds_b128 %0, %1, off"
                 :: "v"(lds_off), "v"(g)
                 : "memory");
}

// One K-step of WMMA compute from LDS buffer 'cur'.
__device__ __forceinline__ void wmma_step(const __bf16 (*sA)[TILE_M][LDSS],
                                          const __bf16 (*sB)[TILE_N][LDSS],
                                          int cur, int waveM, int waveN,
                                          int l16, int hi, v8f acc[2][4])
{
    // ---- A fragments: CDNA5 16-bit A 16x32 layout ----
    v16bf afrag[2];
#pragma unroll
    for (int mt = 0; mt < 2; ++mt) {
        const int row = waveM + mt * 16 + l16;
        const int kb  = hi ? 8 : 0;
        v8bf lo = *(const v8bf*)&sA[cur][row][kb];        // K kb..kb+7
        v8bf hh = *(const v8bf*)&sA[cur][row][kb + 16];   // K kb+16..kb+23
        v16bf a;
#pragma unroll
        for (int i = 0; i < 8; ++i) { a[i] = lo[i]; a[i + 8] = hh[i]; }
        afrag[mt] = a;
    }
    // ---- B fragments (B tile stored N-major so K is contiguous) ----
#pragma unroll
    for (int nt = 0; nt < 4; ++nt) {
        const int col = waveN + nt * 16 + l16;
        const int kb  = hi ? 8 : 0;
        v8bf lo = *(const v8bf*)&sB[cur][col][kb];
        v8bf hh = *(const v8bf*)&sB[cur][col][kb + 16];
        v16bf b;
#pragma unroll
        for (int i = 0; i < 8; ++i) { b[i] = lo[i]; b[i + 8] = hh[i]; }
#pragma unroll
        for (int mt = 0; mt < 2; ++mt)
            acc[mt][nt] = __builtin_amdgcn_wmma_f32_16x16x32_bf16(
                false, afrag[mt], false, b, (short)0, acc[mt][nt], false, false);
    }
}

// C(M,N) = A(M,K) * B(N,K)^T, A/B bf16 in HBM/L2, async-copied straight to LDS.
// Triple-buffered; main loop has unconditional prefetch (last K-tile peeled):
// one s_wait_asynccnt + one barrier per K-step, zero per-iteration branches.
// mode 0: f32 store. mode 1: f32 store of softplus(C + bias[col]). mode 2: bf16 store.
__global__ __launch_bounds__(256)
void gemm_bf16_wmma(const __bf16* __restrict__ A, int lda,
                    const __bf16* __restrict__ B, int ldb,
                    void* __restrict__ Cv, int ldc,
                    int M, int N, int K,
                    const float* __restrict__ bias, int mode)
{
    __shared__ __align__(16) __bf16 sA[3][TILE_M][LDSS];
    __shared__ __align__(16) __bf16 sB[3][TILE_N][LDSS];

    const int tid  = threadIdx.x;
    const int wave = tid >> 5;
    const int lane = tid & 31;
    const int l16  = lane & 15;
    const int hi   = (lane >> 4) & 1;

    const int blockM = blockIdx.y * TILE_M;
    const int blockN = blockIdx.x * TILE_N;
    const int waveM  = (wave >> 1) * 32;   // 4 waves along M (32 rows each)
    const int waveN  = (wave & 1) * 64;    // 2 waves along N (64 cols each)

    v8f acc[2][4];
#pragma unroll
    for (int mt = 0; mt < 2; ++mt)
#pragma unroll
        for (int nt = 0; nt < 4; ++nt)
            acc[mt][nt] = (v8f){0.f,0.f,0.f,0.f,0.f,0.f,0.f,0.f};

    // Copy-slot mapping: 2 slots per matrix; slot p covers row p*64 + tid/4,
    // 16B chunk (tid&3) within the 64B row of the K-tile. Rows are clamped
    // (duplicated data for OOB rows is harmless: those C rows/cols are masked).
    const int crow  = tid >> 2;        // 0..63
    const int cchk  = (tid & 3) * 8;   // bf16 element offset of 16B chunk
    const __bf16* gA[2];
    const __bf16* gB[2];
    unsigned ldsA[2], ldsB[2];
    const unsigned sAbase = (unsigned)(size_t)&sA[0][0][0];
    const unsigned sBbase = (unsigned)(size_t)&sB[0][0][0];
#pragma unroll
    for (int p = 0; p < 2; ++p) {
        const int row = p * 64 + crow;
        int gm = blockM + row; if (gm >= M) gm = M - 1;
        int gn = blockN + row; if (gn >= N) gn = N - 1;
        gA[p] = A + (size_t)gm * lda + cchk;
        gB[p] = B + (size_t)gn * ldb + cchk;
        ldsA[p] = sAbase + (unsigned)(row * LDSS + cchk) * 2u;
        ldsB[p] = sBbase + (unsigned)(row * LDSS + cchk) * 2u;
    }

    // Prologue: tile 0 in flight (4 async ops per thread).
#pragma unroll
    for (int p = 0; p < 2; ++p) async_cp16(ldsA[p], gA[p]);
#pragma unroll
    for (int p = 0; p < 2; ++p) async_cp16(ldsB[p], gB[p]);

    // Main loop over all but the last K-tile: unconditional prefetch.
    int cur = 0;
    for (int k0 = 0; k0 < K - TILE_K; k0 += TILE_K) {
        const int kn  = k0 + TILE_K;
        const int nxt = (cur == 2) ? 0 : cur + 1;
        const unsigned nb = (unsigned)(nxt * TBUF_BYTES);
#pragma unroll
        for (int p = 0; p < 2; ++p) async_cp16(ldsA[p] + nb, gA[p] + kn);
#pragma unroll
        for (int p = 0; p < 2; ++p) async_cp16(ldsB[p] + nb, gB[p] + kn);
        asm volatile("s_wait_asynccnt 4" ::: "memory");   // current tile landed
        __syncthreads();                                  // all waves' copies done

        wmma_step(sA, sB, cur, waveM, waveN, l16, hi, acc);
        cur = nxt;
        // One barrier/iter is safe with 3 buffers: the buffer written next
        // iteration was last read two iterations back, across a barrier.
    }

    // Peeled last K-tile: drain ASYNCcnt fully, no prefetch.
    asm volatile("s_wait_asynccnt 0" ::: "memory");
    __syncthreads();
    wmma_step(sA, sB, cur, waveM, waveN, l16, hi, acc);

    // ---- epilogue/store: C/D 16x16 f32 layout (VGPR r -> row r or r+8) ----
    float*  Cf = (float*)Cv;
    __bf16* Cb = (__bf16*)Cv;
#pragma unroll
    for (int mt = 0; mt < 2; ++mt)
#pragma unroll
        for (int nt = 0; nt < 4; ++nt)
#pragma unroll
            for (int r = 0; r < 8; ++r) {
                const int row = blockM + waveM + mt * 16 + r + hi * 8;
                const int col = blockN + waveN + nt * 16 + l16;
                if (row < M && col < N) {
                    float v = acc[mt][nt][r];
                    if (mode == 1) v = softplus_f(v + bias[col]);
                    if (mode == 2) Cb[(size_t)row * ldc + col] = (__bf16)v;
                    else           Cf[(size_t)row * ldc + col] = v;
                }
            }
}

// Streaming fp32 -> bf16 (n multiple of 1024; one float4 per thread).
__global__ __launch_bounds__(256)
void cvt_f32_to_bf16(const float* __restrict__ src, __bf16* __restrict__ dst)
{
    const size_t i = ((size_t)blockIdx.x * 256 + threadIdx.x) * 4;
    const float4 v = *(const float4*)(src + i);
    dst[i + 0] = (__bf16)v.x; dst[i + 1] = (__bf16)v.y;
    dst[i + 2] = (__bf16)v.z; dst[i + 3] = (__bf16)v.w;
}

// Slice ssm[:, 0:256] -> bf16 dt matrix (2048 x 256).
__global__ __launch_bounds__(256)
void cvt_dt_kernel(const float* __restrict__ ssm, __bf16* __restrict__ dtbf)
{
    const int idx = blockIdx.x * 256 + threadIdx.x;
    const int t = idx >> 8;
    const int j = idx & 255;
    dtbf[idx] = (__bf16)ssm[(size_t)t * NSSM + j];
}

// Depthwise causal conv (D_CONV=4) + bias + SiLU. xz bf16 (SEQ, 2*D_INNER).
__global__ __launch_bounds__(256)
void conv_silu_kernel(const __bf16* __restrict__ xz, const float* __restrict__ cw,
                      const float* __restrict__ cb, __bf16* __restrict__ xact)
{
    const int idx = blockIdx.x * 256 + threadIdx.x;
    const int t = idx >> 13;       // / D_INNER
    const int d = idx & (D_INNER - 1);
    float acc = cb[d];
#pragma unroll
    for (int k = 0; k < D_CONV; ++k) {
        const int tt = t + k - (D_CONV - 1);
        const float xv = (tt >= 0) ? (float)xz[(size_t)tt * (2 * D_INNER) + d] : 0.f;
        acc += cw[d * D_CONV + k] * xv;
    }
    xact[(size_t)t * D_INNER + d] = (__bf16)silu_f(acc);
}

// Selective scan: one thread per channel d, fp32 16-state recurrence in regs.
// Fuses +x*D skip and silu(z) gate; writes bf16 for the output GEMM.
__global__ __launch_bounds__(256)
void scan_kernel(const float* __restrict__ delta, const float* __restrict__ ssm,
                 const __bf16* __restrict__ xact, const __bf16* __restrict__ xz,
                 const float* __restrict__ A_log, const float* __restrict__ Dp,
                 __bf16* __restrict__ yfin)
{
    const int d = blockIdx.x * 256 + threadIdx.x;
    __shared__ float sBC[2 * D_STATE];

    float A[D_STATE];
#pragma unroll
    for (int s = 0; s < D_STATE; ++s) A[s] = A_log[(size_t)d * D_STATE + s];
    const float Dpar = Dp[d];

    float h[D_STATE];
#pragma unroll
    for (int s = 0; s < D_STATE; ++s) h[s] = 0.f;

    for (int t = 0; t < SEQ; ++t) {
        if (threadIdx.x < 2 * D_STATE)
            sBC[threadIdx.x] = ssm[(size_t)t * NSSM + DT_RANK + threadIdx.x];
        __syncthreads();

        const float dl = delta[(size_t)t * D_INNER + d];
        const float u  = (float)xact[(size_t)t * D_INNER + d];
        const float du = dl * u;
        float y = 0.f;
#pragma unroll
        for (int s = 0; s < D_STATE; ++s) {
            h[s] = __expf(dl * A[s]) * h[s] + du * sBC[s];
            y += h[s] * sBC[D_STATE + s];
        }
        const float zv = (float)xz[(size_t)t * (2 * D_INNER) + D_INNER + d];
        yfin[(size_t)t * D_INNER + d] = (__bf16)((y + u * Dpar) * silu_f(zv));
        __syncthreads();
    }
}

extern "C" void kernel_launch(void* const* d_in, const int* in_sizes, int n_in,
                              void* d_out, int out_size, void* d_ws, size_t ws_size,
                              hipStream_t stream) {
    (void)in_sizes; (void)n_in; (void)out_size; (void)ws_size;
    const float* hidden  = (const float*)d_in[0];
    const float* W_in    = (const float*)d_in[1];
    const float* conv_w  = (const float*)d_in[2];
    const float* conv_b  = (const float*)d_in[3];
    const float* W_x     = (const float*)d_in[4];
    const float* W_dt    = (const float*)d_in[5];
    const float* dt_bias = (const float*)d_in[6];
    const float* A_log   = (const float*)d_in[7];
    const float* D_param = (const float*)d_in[8];
    const float* W_out   = (const float*)d_in[9];
    float* out = (float*)d_out;

    // ---- workspace carve-up (bytes) ----
    char* ws = (char*)d_ws;
    size_t off = 0;
    auto carve = [&](size_t bytes) { char* p = ws + off; off += (bytes + 255) & ~(size_t)255; return p; };
    __bf16* xz_bf   = (__bf16*)carve((size_t)SEQ * 2 * D_INNER * 2);      //  67 MB
    __bf16* xact_bf = (__bf16*)carve((size_t)SEQ * D_INNER * 2);          //  33 MB
    float*  ssm     = (float* )carve((size_t)SEQ * NSSM * 4);             // 2.3 MB
    float*  delta   = (float* )carve((size_t)SEQ * D_INNER * 4);          //  67 MB
    __bf16* yfin_bf = (__bf16*)carve((size_t)SEQ * D_INNER * 2);          //  33 MB
    __bf16* hid_bf  = (__bf16*)carve((size_t)SEQ * D_MODEL * 2);          //  17 MB
    __bf16* Win_bf  = (__bf16*)carve((size_t)2 * D_INNER * D_MODEL * 2);  // 134 MB (fits L2)
    __bf16* Wx_bf   = (__bf16*)carve((size_t)NSSM * D_INNER * 2);         // 4.7 MB
    __bf16* Wdt_bf  = (__bf16*)carve((size_t)D_INNER * DT_RANK * 2);      // 4.2 MB
    __bf16* Wout_bf = (__bf16*)carve((size_t)D_MODEL * D_INNER * 2);      //  67 MB (fits L2)
    __bf16* dt_bf   = (__bf16*)carve((size_t)SEQ * DT_RANK * 2);          //   1 MB

    dim3 blk(256);

    // 0) one-shot fp32->bf16 conversions (streamed once; weights then L2-resident)
    cvt_f32_to_bf16<<<(SEQ * D_MODEL) / 1024, blk, 0, stream>>>(hidden, hid_bf);
    cvt_f32_to_bf16<<<(2 * D_INNER * D_MODEL) / 1024, blk, 0, stream>>>(W_in, Win_bf);
    cvt_f32_to_bf16<<<(NSSM * D_INNER) / 1024, blk, 0, stream>>>(W_x, Wx_bf);
    cvt_f32_to_bf16<<<(D_INNER * DT_RANK) / 1024, blk, 0, stream>>>(W_dt, Wdt_bf);
    cvt_f32_to_bf16<<<(D_MODEL * D_INNER) / 1024, blk, 0, stream>>>(W_out, Wout_bf);

    // 1) xz = hidden @ W_in^T   (2048 x 16384, K=4096), bf16 store
    gemm_bf16_wmma<<<dim3(2 * D_INNER / TILE_N, SEQ / TILE_M), blk, 0, stream>>>(
        hid_bf, D_MODEL, Win_bf, D_MODEL, xz_bf, 2 * D_INNER,
        SEQ, 2 * D_INNER, D_MODEL, nullptr, 2);

    // 2) depthwise conv + SiLU -> xact (bf16)
    conv_silu_kernel<<<(SEQ * D_INNER) / 256, blk, 0, stream>>>(xz_bf, conv_w, conv_b, xact_bf);

    // 3) ssm = xact @ W_x^T     (2048 x 288, K=8192), f32 store
    gemm_bf16_wmma<<<dim3((NSSM + TILE_N - 1) / TILE_N, SEQ / TILE_M), blk, 0, stream>>>(
        xact_bf, D_INNER, Wx_bf, D_INNER, ssm, NSSM,
        SEQ, NSSM, D_INNER, nullptr, 0);

    // 3b) dt slice -> bf16
    cvt_dt_kernel<<<(SEQ * DT_RANK) / 256, blk, 0, stream>>>(ssm, dt_bf);

    // 4) delta = softplus(dt @ W_dt^T + dt_bias)   (2048 x 8192, K=256)
    gemm_bf16_wmma<<<dim3(D_INNER / TILE_N, SEQ / TILE_M), blk, 0, stream>>>(
        dt_bf, DT_RANK, Wdt_bf, DT_RANK, delta, D_INNER,
        SEQ, D_INNER, DT_RANK, dt_bias, 1);

    // 5) selective scan + skip + gate -> yfin (bf16)
    scan_kernel<<<D_INNER / 256, blk, 0, stream>>>(delta, ssm, xact_bf, xz_bf,
                                                   A_log, D_param, yfin_bf);

    // 6) out = yfin @ W_out^T   (2048 x 4096, K=8192), f32 store
    gemm_bf16_wmma<<<dim3(D_MODEL / TILE_N, SEQ / TILE_M), blk, 0, stream>>>(
        yfin_bf, D_INNER, Wout_bf, D_INNER, out, D_MODEL,
        SEQ, D_MODEL, D_INNER, nullptr, 0);
}
